// Tight_CLoss_49924699848801
// MI455X (gfx1250) — compile-verified, matrix operand order checked by reading, not run on hardware
//
#include <hip/hip_runtime.h>
#include <stdint.h>

// ---------------------------------------------------------------------------
// MI455X (gfx1250) implementation.
// Stage 1 (bandwidth-bound, ~5.8us floor @23.3TB/s for 128MiB):
//   persistent waves, per-wave double-buffered async global->LDS copies
//   (ASYNCcnt), row losses computed from LDS; logsumexp row-sum done with
//   v_wmma_f32_16x16x32_f16 against a ones matrix (exp values <= 1 -> f16 ok).
// Stage 2 (selection): losses >= 0 => sorted-prefix selection is monotone;
//   coarse histogram -> crossing bin -> fine histogram -> analytic solve.
// ---------------------------------------------------------------------------

typedef _Float16 v16h __attribute__((ext_vector_type(16)));
typedef float    v8f  __attribute__((ext_vector_type(8)));

#define NB   4096           // coarse histogram bins
#define FNB  4096           // fine histogram bins
#define TILE_ROWS 16
#define ROW_BYTES 512       // C=128 * 4B
#define ROW_STRIDE 528      // padded LDS row stride (132 dwords -> bank spread)
#define WAVES_PER_BLOCK 2
#define K1_BLOCKS 2048
#define THRESH_P1 1001.0f   // cum <= 1000 - i  <=>  cum + cnt <= 1001

struct WsHdr {
    unsigned maxbits;     // atomicMax of loss bits (loss >= 0 -> uint order ok)
    unsigned bstar;       // coarse crossing bin (NB = no crossing)
    unsigned cnt_before;  // exact prefix count before bstar
    float    sum_before;  // exact prefix sum before bstar
    float    lo;          // lower edge of bstar
    float    bw;          // coarse bin width
    unsigned pad[10];
};
// ws layout: WsHdr(64B) | ccnt[NB] u32 | csum[NB] f32 | fcnt[FNB] u32 |
//            fsum[FNB] f32 | l[n] f32

// ---------------------------------------------------------------------------
// K0: zero header + histograms (ws is poisoned and never re-poisoned).
// ---------------------------------------------------------------------------
__global__ void init_ws(unsigned* __restrict__ wsu, int ndwords) {
    int i = blockIdx.x * blockDim.x + threadIdx.x;
    int stride = gridDim.x * blockDim.x;
    for (; i < ndwords; i += stride) wsu[i] = 0u;
}

// ---------------------------------------------------------------------------
// Async fill: one 16-row tile (8KB, contiguous in global) into padded LDS.
// Each lane issues 16 x global_load_async_to_lds_b128 (ASYNCcnt += 16/wave).
// ---------------------------------------------------------------------------
__device__ __forceinline__ void issue_fill(unsigned ldsbase,
                                           const float* __restrict__ src,
                                           int tile, int lane) {
    const unsigned tile_byte = (unsigned)tile * (TILE_ROWS * ROW_BYTES);
#pragma unroll
    for (int k = 0; k < 16; ++k) {
        unsigned idx  = (unsigned)(k * 32 + lane);   // 16B-chunk index in tile
        unsigned byte = idx * 16u;
        unsigned row  = byte >> 9;                   // /512
        unsigned col  = byte & 511u;
        unsigned ldsa = ldsbase + row * ROW_STRIDE + col;
        unsigned voff = tile_byte + byte;            // < 128MiB, fits u32
        asm volatile("global_load_async_to_lds_b128 %0, %1, %2"
                     :: "v"(ldsa), "v"(voff), "s"(src)
                     : "memory");
    }
}

#define UPD(xv, cc)                                              \
    do {                                                         \
        float _v = (xv); int _c = (cc);                          \
        m_all = fmaxf(m_all, _v);                                \
        bool _it = (_c == tg);                                   \
        m_oth = fmaxf(m_oth, _it ? -__builtin_inff() : _v);      \
        if (_it) t_sc = _v;                                      \
    } while (0)

// ---------------------------------------------------------------------------
// K1: per-row loss. One wave per 16-row tile; double-buffered async LDS.
// ---------------------------------------------------------------------------
__global__ __launch_bounds__(64) void loss_kernel(
    const float* __restrict__ x, const int* __restrict__ tgt,
    float* __restrict__ lbuf, WsHdr* __restrict__ hdr,
    int ntiles, int nwaves)
{
    __shared__ __align__(16) char tiles[WAVES_PER_BLOCK][2][TILE_ROWS * ROW_STRIDE];
    __shared__ float sred[WAVES_PER_BLOCK][16];

    const int lane = threadIdx.x & 31;
    const int w    = threadIdx.x >> 5;
    const int wave_gid = blockIdx.x * WAVES_PER_BLOCK + w;
    const int r    = lane & 15;       // row within tile owned by this lane
    const bool hiH = lane >= 16;      // high half covers K {8..15,24..31}

    v16h ones;
#pragma unroll
    for (int i = 0; i < 16; ++i) ones[i] = (_Float16)1.0f;

    unsigned ldsbase[2];
    ldsbase[0] = (unsigned)(uintptr_t)(&tiles[w][0][0]);
    ldsbase[1] = (unsigned)(uintptr_t)(&tiles[w][1][0]);

    int t = wave_gid;
    if (t < ntiles) issue_fill(ldsbase[0], x, t, lane);

    int cur = 0;
    while (t < ntiles) {
        const int tn = t + nwaves;
        if (tn < ntiles) {
            issue_fill(ldsbase[cur ^ 1], x, tn, lane);   // prefetch next tile
            asm volatile("s_wait_asynccnt 0x10" ::: "memory"); // cur tile done
        } else {
            asm volatile("s_wait_asynccnt 0x0" ::: "memory");
        }

        const int grow = t * 16 + r;
        const int tg   = tgt[grow];
        const char* rowp = &tiles[w][cur][0] + r * ROW_STRIDE;

        // ---- pass 1: row max / masked max / target score over my 64 cols
        float m_all = -__builtin_inff();
        float m_oth = -__builtin_inff();
        float t_sc  = -__builtin_inff();
#pragma unroll
        for (int c = 0; c < 4; ++c) {
            const int cb = c * 32 + (hiH ? 8 : 0);
            float4 a = *(const float4*)(rowp + (cb +  0) * 4);
            float4 b = *(const float4*)(rowp + (cb +  4) * 4);
            float4 e = *(const float4*)(rowp + (cb + 16) * 4);
            float4 f = *(const float4*)(rowp + (cb + 20) * 4);
            UPD(a.x, cb + 0);  UPD(a.y, cb + 1);  UPD(a.z, cb + 2);  UPD(a.w, cb + 3);
            UPD(b.x, cb + 4);  UPD(b.y, cb + 5);  UPD(b.z, cb + 6);  UPD(b.w, cb + 7);
            UPD(e.x, cb + 16); UPD(e.y, cb + 17); UPD(e.z, cb + 18); UPD(e.w, cb + 19);
            UPD(f.x, cb + 20); UPD(f.y, cb + 21); UPD(f.z, cb + 22); UPD(f.w, cb + 23);
        }
        m_all = fmaxf(m_all, __shfl_xor(m_all, 16, 32));
        m_oth = fmaxf(m_oth, __shfl_xor(m_oth, 16, 32));
        t_sc  = fmaxf(t_sc,  __shfl_xor(t_sc,  16, 32));

        // ---- pass 2: sum(exp(x - max)) via WMMA against ones (K=128 = 4x32)
        v8f acc = {0.f, 0.f, 0.f, 0.f, 0.f, 0.f, 0.f, 0.f};
#pragma unroll
        for (int c = 0; c < 4; ++c) {
            const int cb = c * 32 + (hiH ? 8 : 0);
            float4 a = *(const float4*)(rowp + (cb +  0) * 4);
            float4 b = *(const float4*)(rowp + (cb +  4) * 4);
            float4 e = *(const float4*)(rowp + (cb + 16) * 4);
            float4 f = *(const float4*)(rowp + (cb + 20) * 4);
            v16h A;
            A[0]  = (_Float16)__expf(a.x - m_all);
            A[1]  = (_Float16)__expf(a.y - m_all);
            A[2]  = (_Float16)__expf(a.z - m_all);
            A[3]  = (_Float16)__expf(a.w - m_all);
            A[4]  = (_Float16)__expf(b.x - m_all);
            A[5]  = (_Float16)__expf(b.y - m_all);
            A[6]  = (_Float16)__expf(b.z - m_all);
            A[7]  = (_Float16)__expf(b.w - m_all);
            A[8]  = (_Float16)__expf(e.x - m_all);
            A[9]  = (_Float16)__expf(e.y - m_all);
            A[10] = (_Float16)__expf(e.z - m_all);
            A[11] = (_Float16)__expf(e.w - m_all);
            A[12] = (_Float16)__expf(f.x - m_all);
            A[13] = (_Float16)__expf(f.y - m_all);
            A[14] = (_Float16)__expf(f.z - m_all);
            A[15] = (_Float16)__expf(f.w - m_all);
            acc = __builtin_amdgcn_wmma_f32_16x16x32_f16(
                false, A, false, ones, (short)0, acc, false, false);
        }

        // D layout: lanes 0-15 hold rows 0-7 in acc[0..7]; lanes 16-31 rows 8-15.
        if (lane == 0) {
            sred[w][0] = acc[0]; sred[w][1] = acc[1]; sred[w][2] = acc[2];
            sred[w][3] = acc[3]; sred[w][4] = acc[4]; sred[w][5] = acc[5];
            sred[w][6] = acc[6]; sred[w][7] = acc[7];
        }
        if (lane == 16) {
            sred[w][8]  = acc[0]; sred[w][9]  = acc[1]; sred[w][10] = acc[2];
            sred[w][11] = acc[3]; sred[w][12] = acc[4]; sred[w][13] = acc[5];
            sred[w][14] = acc[6]; sred[w][15] = acc[7];
        }
        asm volatile("s_wait_dscnt 0x0" ::: "memory");
        __builtin_amdgcn_wave_barrier();
        const float se = sred[w][r];

        const float lse    = __logf(se) + m_all;
        const float margin = t_sc - m_oth;
        const float l = (margin >= 0.f) ? fmaxf(0.f, 1.f - margin)
                                        : fmaxf(0.f, 1.f - t_sc + lse);
        if (!hiH) lbuf[grow] = l;

        float lm = l;
        lm = fmaxf(lm, __shfl_xor(lm, 1, 32));
        lm = fmaxf(lm, __shfl_xor(lm, 2, 32));
        lm = fmaxf(lm, __shfl_xor(lm, 4, 32));
        lm = fmaxf(lm, __shfl_xor(lm, 8, 32));
        lm = fmaxf(lm, __shfl_xor(lm, 16, 32));
        if (lane == 0) atomicMax(&hdr->maxbits, __float_as_uint(lm));

        cur ^= 1;
        t = tn;
    }
}

// ---------------------------------------------------------------------------
// K2: coarse histogram (count + sum per bin), LDS-local then global merge.
// ---------------------------------------------------------------------------
__global__ __launch_bounds__(256) void hist_coarse(
    const float* __restrict__ lv, int n, const WsHdr* __restrict__ hdr,
    unsigned* __restrict__ gcnt, float* __restrict__ gsum)
{
    __shared__ unsigned scnt[NB];
    __shared__ float    ssum[NB];
    for (int i = threadIdx.x; i < NB; i += blockDim.x) { scnt[i] = 0u; ssum[i] = 0.f; }
    __syncthreads();

    const float mx = __uint_as_float(hdr->maxbits);
    const float scale = (mx > 0.f) ? ((float)NB / mx) : 0.f;
    int i = blockIdx.x * blockDim.x + threadIdx.x;
    const int stride = gridDim.x * blockDim.x;
    for (; i < n; i += stride) {
        const float v = lv[i];
        int b = (int)(v * scale);
        b = b < 0 ? 0 : (b > NB - 1 ? NB - 1 : b);
        atomicAdd(&scnt[b], 1u);
        atomicAdd(&ssum[b], v);
    }
    __syncthreads();
    for (int j = threadIdx.x; j < NB; j += blockDim.x)
        if (scnt[j]) { atomicAdd(&gcnt[j], scnt[j]); atomicAdd(&gsum[j], ssum[j]); }
}

// ---------------------------------------------------------------------------
// K3: scan coarse bins, find first bin where prefix_sum + prefix_cnt > 1001.
// ---------------------------------------------------------------------------
__global__ __launch_bounds__(256) void scan_coarse(
    WsHdr* __restrict__ hdr, const unsigned* __restrict__ gcnt,
    const float* __restrict__ gsum)
{
    __shared__ unsigned scnt[NB];
    __shared__ float    ssum[NB];
    __shared__ unsigned pcs[256];
    __shared__ float    pss[256];
    __shared__ unsigned sb;
    const int t = threadIdx.x;
    for (int i = t; i < NB; i += 256) { scnt[i] = gcnt[i]; ssum[i] = gsum[i]; }
    if (t == 0) sb = NB;
    __syncthreads();

    unsigned tc = 0; float ts = 0.f;
    for (int j = 0; j < NB / 256; ++j) { int b = t * (NB / 256) + j; tc += scnt[b]; ts += ssum[b]; }
    pcs[t] = tc; pss[t] = ts;
    __syncthreads();
    if (t == 0) {  // exclusive scan of 256 partials
        unsigned rc = 0; float rs = 0.f;
        for (int i = 0; i < 256; ++i) {
            unsigned c = pcs[i]; float s = pss[i];
            pcs[i] = rc; pss[i] = rs; rc += c; rs += s;
        }
    }
    __syncthreads();

    unsigned c0 = pcs[t]; float s0 = pss[t];
    unsigned myb = NB, myc = 0; float mys = 0.f;
    for (int j = 0; j < NB / 256; ++j) {
        int b = t * (NB / 256) + j;
        unsigned c1 = c0 + scnt[b]; float s1 = s0 + ssum[b];
        if (s1 + (float)c1 > THRESH_P1) { myb = (unsigned)b; myc = c0; mys = s0; break; }
        c0 = c1; s0 = s1;
    }
    if (myb < NB) atomicMin(&sb, myb);
    __syncthreads();

    const float mx = __uint_as_float(hdr->maxbits);
    const float bw = (mx > 0.f) ? (mx / (float)NB) : 0.f;
    if (myb == sb && myb < NB) {
        hdr->bstar = myb; hdr->cnt_before = myc; hdr->sum_before = mys;
        hdr->lo = bw * (float)myb; hdr->bw = bw;
    }
    if (t == 255 && sb == NB) {   // no crossing: everything selected
        hdr->bstar = NB; hdr->cnt_before = c0; hdr->sum_before = s0;
        hdr->lo = 0.f; hdr->bw = 0.f;
    }
}

// ---------------------------------------------------------------------------
// K4: fine histogram of values whose coarse bin == bstar.
// ---------------------------------------------------------------------------
__global__ __launch_bounds__(256) void hist_fine(
    const float* __restrict__ lv, int n, const WsHdr* __restrict__ hdr,
    unsigned* __restrict__ gfcnt, float* __restrict__ gfsum)
{
    __shared__ unsigned scnt[FNB];
    __shared__ float    ssum[FNB];
    for (int i = threadIdx.x; i < FNB; i += blockDim.x) { scnt[i] = 0u; ssum[i] = 0.f; }
    __syncthreads();

    const float mx = __uint_as_float(hdr->maxbits);
    const float scale = (mx > 0.f) ? ((float)NB / mx) : 0.f;
    const unsigned bstar = hdr->bstar;
    const float lo = hdr->lo, bw = hdr->bw;
    const float fscale = (bw > 0.f) ? ((float)FNB / bw) : 0.f;

    int i = blockIdx.x * blockDim.x + threadIdx.x;
    const int stride = gridDim.x * blockDim.x;
    for (; i < n; i += stride) {
        const float v = lv[i];
        int b = (int)(v * scale);
        b = b < 0 ? 0 : (b > NB - 1 ? NB - 1 : b);
        if ((unsigned)b == bstar) {
            int fb = (int)((v - lo) * fscale);
            fb = fb < 0 ? 0 : (fb > FNB - 1 ? FNB - 1 : fb);
            atomicAdd(&scnt[fb], 1u);
            atomicAdd(&ssum[fb], v);
        }
    }
    __syncthreads();
    for (int j = threadIdx.x; j < FNB; j += blockDim.x)
        if (scnt[j]) { atomicAdd(&gfcnt[j], scnt[j]); atomicAdd(&gfsum[j], ssum[j]); }
}

// ---------------------------------------------------------------------------
// K5: scan fine bins from exact prefix; analytic within-bin solve; answer.
// ---------------------------------------------------------------------------
__global__ __launch_bounds__(256) void scan_fine(
    const WsHdr* __restrict__ hdr, const unsigned* __restrict__ gfcnt,
    const float* __restrict__ gfsum, float* __restrict__ dout, int n)
{
    __shared__ unsigned scnt[FNB];
    __shared__ float    ssum[FNB];
    __shared__ unsigned pcs[256];
    __shared__ float    pss[256];
    __shared__ unsigned sb;
    const int t = threadIdx.x;
    for (int i = t; i < FNB; i += 256) { scnt[i] = gfcnt[i]; ssum[i] = gfsum[i]; }
    if (t == 0) sb = FNB;
    __syncthreads();

    unsigned tc = 0; float ts = 0.f;
    for (int j = 0; j < FNB / 256; ++j) { int b = t * (FNB / 256) + j; tc += scnt[b]; ts += ssum[b]; }
    pcs[t] = tc; pss[t] = ts;
    __syncthreads();
    if (t == 0) {
        unsigned rc = 0; float rs = 0.f;
        for (int i = 0; i < 256; ++i) {
            unsigned c = pcs[i]; float s = pss[i];
            pcs[i] = rc; pss[i] = rs; rc += c; rs += s;
        }
    }
    __syncthreads();

    const unsigned base_c = hdr->cnt_before;
    const float    base_s = hdr->sum_before;
    unsigned c0 = base_c + pcs[t]; float s0 = base_s + pss[t];
    unsigned myb = FNB, myc = 0; float mys = 0.f;
    for (int j = 0; j < FNB / 256; ++j) {
        int b = t * (FNB / 256) + j;
        unsigned c1 = c0 + scnt[b]; float s1 = s0 + ssum[b];
        if (s1 + (float)c1 > THRESH_P1) { myb = (unsigned)b; myc = c0; mys = s0; break; }
        c0 = c1; s0 = s1;
    }
    if (myb < FNB) atomicMin(&sb, myb);
    __syncthreads();

    if (myb == sb && myb < FNB) {
        const unsigned fc = scnt[myb];
        const float    fs = ssum[myb];
        const float v = fs / (float)fc;                 // crossing bin is non-empty
        const float avail = THRESH_P1 - (float)myc - mys;
        float jf = floorf(avail / (v + 1.f));
        jf = jf < 0.f ? 0.f : jf;
        jf = jf > (float)fc ? (float)fc : jf;
        const float k     = (float)myc + jf;
        const float loss1 = mys + jf * v;
        const float loss2 = (float)n - k;
        dout[0] = (loss1 < loss2) ? loss2 : loss1;
    }
    if (t == 255 && sb == FNB) {                        // all selected
        const float loss1 = s0;
        const float loss2 = (float)n - (float)c0;
        dout[0] = (loss1 < loss2) ? loss2 : loss1;
    }
}

// ---------------------------------------------------------------------------
extern "C" void kernel_launch(void* const* d_in, const int* in_sizes, int n_in,
                              void* d_out, int out_size, void* d_ws, size_t ws_size,
                              hipStream_t stream) {
    (void)n_in; (void)out_size; (void)ws_size;
    const float* x  = (const float*)d_in[0];
    const int*   tg = (const int*)d_in[1];
    const int n = in_sizes[1];

    char* ws = (char*)d_ws;
    WsHdr*    hdr  = (WsHdr*)ws;
    unsigned* ccnt = (unsigned*)(ws + 64);
    float*    csum = (float*)   (ws + 64 + NB * 4);
    unsigned* fcnt = (unsigned*)(ws + 64 + NB * 8);
    float*    fsum = (float*)   (ws + 64 + NB * 8 + FNB * 4);
    float*    lbuf = (float*)   (ws + 64 + NB * 8 + FNB * 8);

    const int ndw = (64 + NB * 8 + FNB * 8) / 4;
    init_ws<<<64, 256, 0, stream>>>((unsigned*)ws, ndw);

    const int ntiles = n / TILE_ROWS;
    const int nwaves = K1_BLOCKS * WAVES_PER_BLOCK;
    loss_kernel<<<K1_BLOCKS, 64, 0, stream>>>(x, tg, lbuf, hdr, ntiles, nwaves);

    hist_coarse<<<1024, 256, 0, stream>>>(lbuf, n, hdr, ccnt, csum);
    scan_coarse<<<1, 256, 0, stream>>>(hdr, ccnt, csum);
    hist_fine<<<1024, 256, 0, stream>>>(lbuf, n, hdr, fcnt, fsum);
    scan_fine<<<1, 256, 0, stream>>>(hdr, fcnt, fsum, (float*)d_out, n);
}
